// SGCRN_RD_22325240005088
// MI455X (gfx1250) — matrix operand order repaired and checked by reading, not compiled, not run on hardware
//
#include <hip/hip_runtime.h>

// ---------------- problem constants ----------------
#define Bb   64
#define Tt   12
#define Nn   1024
#define Hh   64
#define CI   65          // D_IN + H
#define Jj   4160        // B * CI
#define KKP  80          // padded (K*CI=130 -> 160) / 2 packed pairs
#define OG   128         // 2*H (gate output)
#define OU   64          // H (up output)

typedef __attribute__((ext_vector_type(16))) __bf16        v16bf;
typedef __attribute__((ext_vector_type(8)))  float         v8f;
typedef __attribute__((ext_vector_type(8)))  unsigned int  v8u;
typedef __attribute__((ext_vector_type(4)))  unsigned int  v4u;
typedef __attribute__((ext_vector_type(8)))  int           v8i;
typedef __attribute__((ext_vector_type(4)))  int           v4i;

union FragU { v8u u; v16bf b; };

static __device__ __forceinline__ unsigned short f2bf(float f) {
  unsigned int u = __float_as_uint(f);
  u += 0x7FFFu + ((u >> 16) & 1u);          // round-to-nearest-even
  return (unsigned short)(u >> 16);
}

static __device__ __forceinline__ v8f wmma_bf16(const FragU &a, const FragU &b, v8f c) {
  return __builtin_amdgcn_wmma_f32_16x16x32_bf16(false, a.b, false, b.b,
                                                 (short)0, c, false, false);
}

// A-matrix fragment pair index within a 16-pair (K=32) chunk.
// VGPR v, half = lane/16:  K_base = (v<4 ? 2v : 16+2(v-4)) + 8*half  -> kk = K_base/2
static __device__ __forceinline__ int a_kk(int v, int half) {
  return (v < 4 ? v : 8 + (v - 4)) + 4 * half;
}

// Pair-interleaved activation layout: XP[m>>1][j] (uint32) packs rows (2mp | 2mp+1<<16).
static __device__ __forceinline__ unsigned int xp_ld(const unsigned short *Xp, int m, int j) {
  return Xp[(((size_t)(m >> 1)) * Jj + j) * 2 + (m & 1)];
}
static __device__ __forceinline__ void xp_st(unsigned short *Xp, int m, int j, unsigned short v) {
  Xp[(((size_t)(m >> 1)) * Jj + j) * 2 + (m & 1)] = v;
}

// ---------------- Tensor Data Mover: 2-D tile (dword elements) global -> LDS ---------------
// Descriptor per CDNA5 ISA ch.8: group0 = {count|flags, lds_addr, gaddr_lo, gaddr_hi|type=2},
// group1 = {data_size=4B, tensor_dim0/1, tile_dim0/1, tensor_dim0_stride}; groups2/3 = 0 (2D).
// This toolchain exposes the 6-arg builtin: (u32x4, i32x8, i32x4, i32x4, i32x8, i32 cpol).
static __device__ __forceinline__ void tdm_load_2d_dw(unsigned lds_addr, const void *gptr,
                                                      unsigned tensor_w, unsigned tensor_h,
                                                      unsigned tile_w, unsigned tile_h,
                                                      unsigned row_stride) {
  unsigned long long ga = (unsigned long long)(__SIZE_TYPE__)gptr;
  v4u g0;
  g0[0] = 1u;                                                 // count=1, user mode
  g0[1] = lds_addr;                                           // LDS byte address
  g0[2] = (unsigned)ga;                                       // global addr [31:0]
  g0[3] = (unsigned)((ga >> 32) & 0x01FFFFFFu) | (2u << 30);  // addr[56:32] | type=2
  v8i g1;
  g1[0] = (int)(2u << 16);                                    // data_size code 2 = 4 bytes
  g1[1] = (int)(tensor_w << 16);                              // tensor_dim0[15:0]
  g1[2] = (int)((tensor_w >> 16) | (tensor_h << 16));         // dim0 hi | tensor_dim1 lo
  g1[3] = (int)((tensor_h >> 16) | (tile_w << 16));           // dim1 hi | tile_dim0
  g1[4] = (int)(tile_h & 0xFFFFu);                            // tile_dim1 (tile_dim2=0)
  g1[5] = (int)row_stride;                                    // tensor_dim0_stride lo32
  g1[6] = 0;
  g1[7] = 0;
  v4i z4 = {0, 0, 0, 0};
  v8i z8 = {0, 0, 0, 0, 0, 0, 0, 0};
  __builtin_amdgcn_tensor_load_to_lds(g0, g1, z4, z4, z8, 0);
}

// ---------------- A = softmax(relu(E E^T)), stored bf16 row-major ----------------
__global__ __launch_bounds__(256) void k_build_A(const float *__restrict__ E,
                                                 unsigned short *__restrict__ Abf) {
  __shared__ float er[16];
  __shared__ float s[Nn];
  __shared__ float red[256];
  const int row = blockIdx.x, tid = threadIdx.x;
  if (tid < 16) er[tid] = E[row * 16 + tid];
  __syncthreads();
  float lmax = -1e30f;
  for (int j = tid; j < Nn; j += 256) {
    const float *ej = E + j * 16;
    float d = 0.f;
#pragma unroll
    for (int t = 0; t < 16; ++t) d += er[t] * ej[t];
    d = fmaxf(d, 0.f);
    s[j] = d;
    lmax = fmaxf(lmax, d);
  }
  red[tid] = lmax; __syncthreads();
  for (int o = 128; o > 0; o >>= 1) { if (tid < o) red[tid] = fmaxf(red[tid], red[tid + o]); __syncthreads(); }
  const float mx = red[0]; __syncthreads();
  float lsum = 0.f;
  for (int j = tid; j < Nn; j += 256) { float e = expf(s[j] - mx); s[j] = e; lsum += e; }
  red[tid] = lsum; __syncthreads();
  for (int o = 128; o > 0; o >>= 1) { if (tid < o) red[tid] += red[tid + o]; __syncthreads(); }
  const float inv = 1.f / red[0];
  for (int j = tid; j < Nn; j += 256) Abf[row * Nn + j] = f2bf(s[j] * inv);
}

// ---------------- per-node weights: Wp[n][kk][o] = pack2(bf16(sum_d E[n,d] W[d,k,i,o])) ----
__global__ __launch_bounds__(256) void k_expand_W(const float *__restrict__ E,
                                                  const float *__restrict__ W,
                                                  unsigned int *__restrict__ Wp, int O) {
  __shared__ float en[16];
  const int n = blockIdx.x, tid = threadIdx.x;
  if (tid < 16) en[tid] = E[n * 16 + tid];
  __syncthreads();
  const int total = KKP * O;
  for (int idx = tid; idx < total; idx += 256) {
    const int kk = idx / O, o = idx % O;
    float v0 = 0.f, v1 = 0.f;
    const int kc0 = 2 * kk, kc1 = 2 * kk + 1;
    if (kc0 < 130) {
      const int k = kc0 / CI, i = kc0 % CI;
#pragma unroll
      for (int d = 0; d < 16; ++d) v0 += en[d] * W[((d * 2 + k) * CI + i) * O + o];
    }
    if (kc1 < 130) {
      const int k = kc1 / CI, i = kc1 % CI;
#pragma unroll
      for (int d = 0; d < 16; ++d) v1 += en[d] * W[((d * 2 + k) * CI + i) * O + o];
    }
    Wp[(size_t)n * total + idx] = (unsigned int)f2bf(v0) | ((unsigned int)f2bf(v1) << 16);
  }
}

__global__ void k_expand_b(const float *__restrict__ E, const float *__restrict__ bsrc,
                           float *__restrict__ bdst, int O) {
  __shared__ float en[16];
  const int n = blockIdx.x, tid = threadIdx.x;
  if (tid < 16) en[tid] = E[n * 16 + tid];
  __syncthreads();
  for (int o = tid; o < O; o += blockDim.x) {
    float v = 0.f;
#pragma unroll
    for (int d = 0; d < 16; ++d) v += en[d] * bsrc[d * O + o];
    bdst[n * O + o] = v;
  }
}

__global__ void k_zero(float *__restrict__ p, int nf) {
  int i = blockIdx.x * 256 + threadIdx.x;
  if (i < nf) p[i] = 0.f;
}

// ---------------- XP[mp][j] <- pack(x_t|h for rows 2mp, 2mp+1) ----------------
__global__ __launch_bounds__(256) void k_concat(const float *__restrict__ src, int t,
                                                const float *__restrict__ h,
                                                unsigned int *__restrict__ Xp) {
  const int mp = blockIdx.x;                       // 0..511
  const int n0 = 2 * mp, n1 = n0 + 1;
  for (int j = threadIdx.x; j < Jj; j += 256) {
    const int b = j / CI, c = j % CI;
    float v0, v1;
    if (c == 0) {
      v0 = src[((size_t)b * Tt + t) * Nn + n0];
      v1 = src[((size_t)b * Tt + t) * Nn + n1];
    } else {
      v0 = h[((size_t)n0 * Bb + b) * Hh + (c - 1)];
      v1 = h[((size_t)n1 * Bb + b) * Hh + (c - 1)];
    }
    Xp[(size_t)mp * Jj + j] = (unsigned)f2bf(v0) | ((unsigned)f2bf(v1) << 16);
  }
}

__global__ void k_setxt(const float *__restrict__ src, int t, unsigned int *__restrict__ Xp) {
  const int mp = blockIdx.x, b = threadIdx.x;      // blockDim = 64
  const int n0 = 2 * mp, n1 = n0 + 1;
  const unsigned lo = f2bf(src[((size_t)b * Tt + t) * Nn + n0]);
  const unsigned hi = f2bf(src[((size_t)b * Tt + t) * Nn + n1]);
  Xp[(size_t)mp * Jj + b * CI] = lo | (hi << 16);
}

// ---------------- big WMMA GEMM: Z[n][j] = sum_m A[n][m] X[m][j] ----------------
// TDM stages both panels; grid = (Jj/64, Nn/128); 8 waves, wave w -> 16 rows x 64 cols
__global__ __launch_bounds__(256) void k_diffusion(const unsigned short *__restrict__ Abf,
                                                   const unsigned int *__restrict__ Xp,
                                                   unsigned short *__restrict__ Zp) {
  __shared__ unsigned int lsA[128 * 16];   // 128 rows x 16 K-pairs (dwords)
  __shared__ unsigned int lsB[16 * 64];    // 16 K-pair rows x 64 cols (dwords)
  const int tid = threadIdx.x;
  const int lane = tid & 31;
  const int wv = __builtin_amdgcn_readfirstlane(tid) >> 5;   // scalar wave id
  const int wave = tid >> 5;
  const int half = lane >> 4, q = lane & 15;
  const int n0 = blockIdx.y * 128, j0 = blockIdx.x * 64;
  const unsigned lsA_addr = (unsigned)(__SIZE_TYPE__)&lsA[0];
  const unsigned lsB_addr = (unsigned)(__SIZE_TYPE__)&lsB[0];
  const unsigned int *Adw = (const unsigned int *)Abf;       // A as [1024][512] dwords
  v8f zero = {};
  v8f acc[4] = {zero, zero, zero, zero};
  for (int kc = 0; kc < 32; ++kc) {                          // K = 1024 = 32 chunks of 32
    if (wv == 0) {
      // A panel: rows n0..n0+127, dword cols kc*16..+16  -> lsA[r][kk]
      tdm_load_2d_dw(lsA_addr, Adw + (size_t)n0 * 512 + kc * 16,
                     512u, 1024u, 16u, 128u, 512u);
      // B panel: pair-rows kc*16..+16, cols j0..j0+64     -> lsB[kk][c]
      tdm_load_2d_dw(lsB_addr, Xp + (size_t)(kc * 16) * Jj + j0,
                     (unsigned)Jj, 512u, 64u, 16u, (unsigned)Jj);
      __builtin_amdgcn_s_wait_tensorcnt(0);
    }
    __syncthreads();
    FragU af;
#pragma unroll
    for (int v = 0; v < 8; ++v) af.u[v] = lsA[(wave * 16 + q) * 16 + v + 8 * half];
#pragma unroll
    for (int nt = 0; nt < 4; ++nt) {
      FragU bf;
#pragma unroll
      for (int v = 0; v < 8; ++v) bf.u[v] = lsB[(v + 8 * half) * 64 + nt * 16 + q];
      acc[nt] = wmma_bf16(af, bf, acc[nt]);
    }
    __syncthreads();
  }
  unsigned short *Zus = Zp;
#pragma unroll
  for (int nt = 0; nt < 4; ++nt)
#pragma unroll
    for (int v = 0; v < 8; ++v) {
      const int n = n0 + wave * 16 + v + 8 * half;
      const int j = j0 + nt * 16 + q;
      xp_st(Zus, n, j, f2bf(acc[nt][v]));
    }
}

// zg element (bf16 in pair-interleaved memory): kc<65 identity part, kc<130 A@xs part, else 0
static __device__ __forceinline__ unsigned int zg_elem(const unsigned short *X,
                                                       const unsigned short *Z1,
                                                       int n, int b, int kc) {
  if (kc < CI)  return xp_ld(X, n, b * CI + kc);
  if (kc < 130) return xp_ld(Z1, n, b * CI + (kc - CI));
  return 0u;
}

// ---------------- per-node gate GEMM + sigmoid + (z*h -> Xc cols 1..64, r -> rbuf) ----------
__global__ __launch_bounds__(256) void k_gate(const unsigned short *__restrict__ Xbf,
                                              const unsigned short *__restrict__ Z1,
                                              const unsigned int *__restrict__ Wp,
                                              const float *__restrict__ bg,
                                              const float *__restrict__ h,
                                              float *__restrict__ rbuf,
                                              unsigned short *__restrict__ Xc) {
  __shared__ unsigned int lsZ[Bb * KKP];             // 64 x 80 pairs = 20 KB
  const int n = blockIdx.x, tid = threadIdx.x;
  const int lane = tid & 31, wave = tid >> 5;
  const int half = lane >> 4, q = lane & 15;
  const int m_tile = wave & 3, o_half = wave >> 2;   // 4 b-tiles x 2 o-halves
  for (int idx = tid; idx < Bb * KKP; idx += 256) {
    const int b = idx / KKP, kk = idx % KKP;
    lsZ[idx] = zg_elem(Xbf, Z1, n, b, 2 * kk) | (zg_elem(Xbf, Z1, n, b, 2 * kk + 1) << 16);
  }
  __syncthreads();
  v8f zero = {};
  v8f acc[4] = {zero, zero, zero, zero};
  const unsigned int *Wn = Wp + (size_t)n * (KKP * OG);
  for (int kc = 0; kc < 5; ++kc) {                   // Kpad = 160 = 5 chunks of 32
    if (kc + 1 < 5) __builtin_prefetch(&Wn[(size_t)((kc + 1) * 16) * OG], 0, 1);
    FragU af;
#pragma unroll
    for (int v = 0; v < 8; ++v)
      af.u[v] = lsZ[(m_tile * 16 + q) * KKP + kc * 16 + v + 8 * half];
#pragma unroll
    for (int nt = 0; nt < 4; ++nt) {
      FragU bf;
      const int c = o_half * 64 + nt * 16 + q;
#pragma unroll
      for (int v = 0; v < 8; ++v) bf.u[v] = Wn[(size_t)(kc * 16 + v + 8 * half) * OG + c];
      acc[nt] = wmma_bf16(af, bf, acc[nt]);
    }
  }
#pragma unroll
  for (int nt = 0; nt < 4; ++nt)
#pragma unroll
    for (int v = 0; v < 8; ++v) {
      const int b = m_tile * 16 + v + 8 * half;
      const int o = o_half * 64 + nt * 16 + q;
      const float x = acc[nt][v] + bg[n * OG + o];
      const float s = 1.0f / (1.0f + expf(-x));
      if (o < Hh) {
        const float zh = s * h[((size_t)n * Bb + b) * Hh + o];
        xp_st(Xc, n, b * CI + 1 + o, f2bf(zh));
      } else {
        rbuf[((size_t)n * Bb + b) * Hh + (o - Hh)] = s;
      }
    }
}

// ---------------- per-node candidate GEMM + tanh + GRU update (h in place) ----------------
__global__ __launch_bounds__(256) void k_up(const unsigned short *__restrict__ Xc,
                                            const unsigned short *__restrict__ Zc1,
                                            const unsigned int *__restrict__ Wp,
                                            const float *__restrict__ bu,
                                            const float *__restrict__ rbuf,
                                            float *__restrict__ h) {
  __shared__ unsigned int lsZ[Bb * KKP];
  const int n = blockIdx.x, tid = threadIdx.x;
  const int lane = tid & 31, wave = tid >> 5;
  const int half = lane >> 4, q = lane & 15;
  const int m_tile = wave & 3, o_pair = wave >> 2;   // 4 b-tiles x 2 o-pairs (2 tiles each)
  for (int idx = tid; idx < Bb * KKP; idx += 256) {
    const int b = idx / KKP, kk = idx % KKP;
    lsZ[idx] = zg_elem(Xc, Zc1, n, b, 2 * kk) | (zg_elem(Xc, Zc1, n, b, 2 * kk + 1) << 16);
  }
  __syncthreads();
  v8f zero = {};
  v8f acc[2] = {zero, zero};
  const unsigned int *Wn = Wp + (size_t)n * (KKP * OU);
  for (int kc = 0; kc < 5; ++kc) {
    if (kc + 1 < 5) __builtin_prefetch(&Wn[(size_t)((kc + 1) * 16) * OU], 0, 1);
    FragU af;
#pragma unroll
    for (int v = 0; v < 8; ++v)
      af.u[v] = lsZ[(m_tile * 16 + q) * KKP + kc * 16 + v + 8 * half];
#pragma unroll
    for (int nt = 0; nt < 2; ++nt) {
      FragU bf;
      const int c = o_pair * 32 + nt * 16 + q;
#pragma unroll
      for (int v = 0; v < 8; ++v) bf.u[v] = Wn[(size_t)(kc * 16 + v + 8 * half) * OU + c];
      acc[nt] = wmma_bf16(af, bf, acc[nt]);
    }
  }
#pragma unroll
  for (int nt = 0; nt < 2; ++nt)
#pragma unroll
    for (int v = 0; v < 8; ++v) {
      const int b = m_tile * 16 + v + 8 * half;
      const int o = o_pair * 32 + nt * 16 + q;
      const float hc = tanhf(acc[nt][v] + bu[n * OU + o]);
      const size_t i = ((size_t)n * Bb + b) * Hh + o;
      const float r = rbuf[i], ho = h[i];
      h[i] = r * ho + (1.f - r) * hc;                // each (n,b,o) owned by one lane
    }
}

// ---------------- output heads ----------------
__global__ void k_proj1(const float *__restrict__ h, const float *__restrict__ c1W,
                        const float *__restrict__ c1b, const float *__restrict__ c2W,
                        const float *__restrict__ c2b, const float *__restrict__ source,
                        float *__restrict__ out1, float *__restrict__ src2) {
  const int idx = blockIdx.x * blockDim.x + threadIdx.x;
  const int n = idx & (Nn - 1), b = idx >> 10;
  if (b >= Bb) return;
  float hv[Hh];
#pragma unroll
  for (int i = 0; i < Hh; ++i) hv[i] = h[((size_t)n * Bb + b) * Hh + i];
  float o1[Tt];
  for (int c = 0; c < Tt; ++c) {
    float s = c1b[c];
#pragma unroll
    for (int i = 0; i < Hh; ++i) s += hv[i] * c1W[c * Hh + i];
    o1[c] = s;
    out1[((size_t)b * Tt + c) * Nn + n] = s;
  }
  for (int c = 0; c < Tt; ++c) {
    float s = c2b[c];
#pragma unroll
    for (int j = 0; j < Tt; ++j) s += o1[j] * c2W[c * Tt + j];
    src2[((size_t)b * Tt + c) * Nn + n] = source[((size_t)b * Tt + c) * Nn + n] - s;
  }
}

__global__ void k_proj2(const float *__restrict__ h, const float *__restrict__ c3W,
                        const float *__restrict__ c3b, const float *__restrict__ out1,
                        float *__restrict__ out) {
  const int idx = blockIdx.x * blockDim.x + threadIdx.x;
  const int n = idx & (Nn - 1), b = idx >> 10;
  if (b >= Bb) return;
  float hv[Hh];
#pragma unroll
  for (int i = 0; i < Hh; ++i) hv[i] = h[((size_t)n * Bb + b) * Hh + i];
  for (int c = 0; c < Tt; ++c) {
    float s = c3b[c];
#pragma unroll
    for (int i = 0; i < Hh; ++i) s += hv[i] * c3W[c * Hh + i];
    out[((size_t)b * Tt + c) * Nn + n] = out1[((size_t)b * Tt + c) * Nn + n] + s;
  }
}

// ---------------- host orchestration ----------------
extern "C" void kernel_launch(void *const *d_in, const int *in_sizes, int n_in,
                              void *d_out, int out_size, void *d_ws, size_t ws_size,
                              hipStream_t stream) {
  (void)in_sizes; (void)n_in; (void)out_size; (void)ws_size;
  const float *source = (const float *)d_in[0];
  const float *Emb    = (const float *)d_in[1];
  const float *gW[2]  = {(const float *)d_in[2], (const float *)d_in[6]};
  const float *gB[2]  = {(const float *)d_in[3], (const float *)d_in[7]};
  const float *uW[2]  = {(const float *)d_in[4], (const float *)d_in[8]};
  const float *uB[2]  = {(const float *)d_in[5], (const float *)d_in[9]};
  const float *c1W = (const float *)d_in[10], *c1b = (const float *)d_in[11];
  const float *c2W = (const float *)d_in[12], *c2b = (const float *)d_in[13];
  const float *c3W = (const float *)d_in[14], *c3b = (const float *)d_in[15];
  float *out = (float *)d_out;

  // workspace carve-out (~140 MB total; encoder-2 weights overlay encoder-1's)
  char *p = (char *)d_ws;
  auto carve = [&](size_t bytes) -> char * {
    char *r = p; p += (bytes + 255) & ~(size_t)255; return r;
  };
  unsigned short *Abf  = (unsigned short *)carve(2ull * Nn * Nn);
  unsigned int   *Wg   = (unsigned int *)carve(4ull * Nn * KKP * OG);
  unsigned int   *Wu   = (unsigned int *)carve(4ull * Nn * KKP * OU);
  float          *bg   = (float *)carve(4ull * Nn * OG);
  float          *bu   = (float *)carve(4ull * Nn * OU);
  unsigned short *Xbf  = (unsigned short *)carve(2ull * Nn * Jj);   // pair-interleaved
  unsigned short *Z1   = (unsigned short *)carve(2ull * Nn * Jj);
  unsigned short *Xc   = (unsigned short *)carve(2ull * Nn * Jj);
  unsigned short *Zc1  = (unsigned short *)carve(2ull * Nn * Jj);
  float          *rbuf = (float *)carve(4ull * Nn * Bb * Hh);
  float          *hb   = (float *)carve(4ull * Nn * Bb * Hh);
  float          *out1 = (float *)carve(4ull * Bb * Tt * Nn);
  float          *src2 = (float *)carve(4ull * Bb * Tt * Nn);

  k_build_A<<<Nn, 256, 0, stream>>>(Emb, Abf);

  for (int e = 0; e < 2; ++e) {
    k_expand_W<<<Nn, 256, 0, stream>>>(Emb, gW[e], Wg, OG);
    k_expand_W<<<Nn, 256, 0, stream>>>(Emb, uW[e], Wu, OU);
    k_expand_b<<<Nn, 128, 0, stream>>>(Emb, gB[e], bg, OG);
    k_expand_b<<<Nn, 64, 0, stream>>>(Emb, uB[e], bu, OU);
    k_zero<<<(Nn * Bb * Hh + 255) / 256, 256, 0, stream>>>(hb, Nn * Bb * Hh);
    const float *sp = (e == 0) ? source : src2;
    for (int t = 0; t < Tt; ++t) {
      k_concat<<<Nn / 2, 256, 0, stream>>>(sp, t, hb, (unsigned int *)Xbf);
      k_diffusion<<<dim3(Jj / 64, Nn / 128), 256, 0, stream>>>(Abf, (const unsigned int *)Xbf, Z1);
      k_setxt<<<Nn / 2, Bb, 0, stream>>>(sp, t, (unsigned int *)Xc);
      k_gate<<<Nn, 256, 0, stream>>>(Xbf, Z1, Wg, bg, hb, rbuf, Xc);
      k_diffusion<<<dim3(Jj / 64, Nn / 128), 256, 0, stream>>>(Abf, (const unsigned int *)Xc, Zc1);
      k_up<<<Nn, 256, 0, stream>>>(Xc, Zc1, Wu, bu, rbuf, hb);
    }
    if (e == 0)
      k_proj1<<<Nn * Bb / 256, 256, 0, stream>>>(hb, c1W, c1b, c2W, c2b, source, out1, src2);
    else
      k_proj2<<<Nn * Bb / 256, 256, 0, stream>>>(hb, c3W, c3b, out1, out);
  }
}
